// MSMT_LE_65343632441647
// MI455X (gfx1250) — matrix operand order repaired and checked by reading, not compiled, not run on hardware
//
#include <hip/hip_runtime.h>
#include <hip/hip_bf16.h>

typedef __attribute__((ext_vector_type(16))) _Float16 v16h;
typedef __attribute__((ext_vector_type(8)))  _Float16 v8h;
typedef __attribute__((ext_vector_type(2)))  _Float16 v2h;
typedef __attribute__((ext_vector_type(8)))  float    v8f;

#define BQ 16                       // rows per wave tile
#define WAVES 2                     // waves per block
#define ROWS_PER_BLOCK (BQ * WAVES) // 32
#define NTH (WAVES * 32)            // 64 threads
#define WBUF_ELEMS (64 * 136)       // 64 cols x (128+8) padded K, f16

#define DEV_INLINE __device__ __attribute__((always_inline)) inline

struct KParams {
    const float* x;
    const float* proj_w;     const float* proj_b;
    const float* qkv_w;      const float* qkv_b;
    const float* attnproj_w; const float* attnproj_b;
    const float* mlp_w1;     const float* mlp_b1;
    const float* mlp_w2;     const float* mlp_b2;
    const float* rel_w1;     const float* rel_b1;
    const float* rel_w2;     const float* rel_b2;
    const float* rel_w3;     const float* rel_b3;
    const float* gate_w1;    const float* gate_b1;
    const float* gate_w2;    const float* gate_b2;
    const float* ctrl_w1;    const float* ctrl_b1;
    const float* ctrl_w2;    const float* ctrl_b2;
    const float* L_w[5];     const float* L_b[5];
    const float* H_w[3];     const float* H_b[3];
    float* out;
};

DEV_INLINE v8f z8() {
    v8f z = {0.f, 0.f, 0.f, 0.f, 0.f, 0.f, 0.f, 0.f};
    return z;
}

// A fragment (16x32 f16, wave32 layout per ISA 7.12.2):
// lane r (0-15): row r, K = {k0+8*hi+0..7} U {k0+16+8*hi+0..7} with hi = lane>>4.
// Both runs are 8 contiguous, 16B-aligned halfs -> 2x ds_load_b128.
DEV_INLINE v16h load_a16(const _Float16* src, int lda, int lane, int k0) {
    int r = lane & 15, hi = (lane >> 4) & 1;
    const v8h* p = (const v8h*)(src + r * lda + k0 + 8 * hi);
    v8h lo = p[0];
    v8h up = p[2];   // +16 halfs
    return __builtin_shufflevector(lo, up, 0, 1, 2, 3, 4, 5, 6, 7,
                                           8, 9, 10, 11, 12, 13, 14, 15);
}

// B fragment (32x16 f16): lane holds column n = lane&15, K = {k0+8*hi+0..7} U
// {k0+16+8*hi+0..7} with hi = lane>>4. Weights are staged COLUMN-MAJOR in LDS
// (wb[n*ldk + k], ldk 16B-aligned) so this is 2x ds_load_b128.
DEV_INLINE v16h load_b16(const _Float16* wb, int ldk, int lane, int k0, int n0) {
    int n = lane & 15, khi = (lane >> 4) & 1;
    const v8h* p = (const v8h*)(wb + (n0 + n) * ldk + k0 + 8 * khi);
    v8h lo = p[0];
    v8h up = p[2];
    return __builtin_shufflevector(lo, up, 0, 1, 2, 3, 4, 5, 6, 7,
                                           8, 9, 10, 11, 12, 13, 14, 15);
}

DEV_INLINE v8f wmma_f16(v16h a, v16h b, v8f c) {
    return __builtin_amdgcn_wmma_f32_16x16x32_f16(false, a, false, b, (short)0, c, false, false);
}

// Cooperatively stage W[kc:kc+kcw, nc:nc+64] (f32 global, row stride N)
// -> wbuf f16 COLUMN-MAJOR [64][ldk], zero padded. Two K-adjacent values are
// packed per thread -> ds_store_b32; global reads stay coalesced over n.
DEV_INLINE void stage_w(_Float16* wbuf, const float* W, int K, int N,
                        int kc, int nc, int kcw, int ldk, int tid) {
    int total = (kcw >> 1) * 64;
    for (int idx = tid; idx < total; idx += NTH) {
        int kp = idx >> 6, n = idx & 63;
        int k = kp * 2;
        int gk = kc + k, gn = nc + n;
        float v0 = (gk < K && gn < N) ? W[(size_t)gk * N + gn] : 0.0f;
        float v1 = (gk + 1 < K && gn < N) ? W[(size_t)(gk + 1) * N + gn] : 0.0f;
        v2h pv = { (_Float16)v0, (_Float16)v1 };
        *(v2h*)(wbuf + n * ldk + k) = pv;
    }
}

// D-layout epilogue: lane holds D[m = i+8*hi][n = lane&15] in acc[nt][i]
DEV_INLINE void epilogue(const v8f* acc, int nc, int N, const float* bias,
                         const _Float16* res, int ldr, int relu,
                         _Float16* dst, int ldd, int lane) {
    int n = lane & 15, hi = (lane >> 4) & 1;
#pragma unroll
    for (int nt = 0; nt < 4; ++nt) {
        int gn = nc + nt * 16 + n;
        float bv = (bias && gn < N) ? bias[gn] : 0.0f;
#pragma unroll
        for (int i = 0; i < 8; ++i) {
            int m = i + 8 * hi;
            float v = acc[nt][i] + bv;
            if (res && gn < N) v += (float)res[m * ldr + gn];
            if (relu) v = fmaxf(v, 0.0f);
            dst[m * ldd + gn] = (gn < N) ? (_Float16)v : (_Float16)0.0f;
        }
    }
}

// GEMM with A fragments preloaded in registers (K <= 128). In-place dst allowed.
DEV_INLINE void gemm_pre(const v16h* afr, int nk, int K, int N,
                         const float* W, const float* bias,
                         const _Float16* res, int ldr, int relu,
                         _Float16* dst, int ldd,
                         _Float16* wbuf, int lane, int tid) {
    int ldk = nk * 32 + 8;
    for (int nc = 0; nc < N; nc += 64) {
        v8f acc[4] = {z8(), z8(), z8(), z8()};
        stage_w(wbuf, W, K, N, 0, nc, nk * 32, ldk, tid);
        __syncthreads();
        for (int ks = 0; ks < nk; ++ks) {
#pragma unroll
            for (int nt = 0; nt < 4; ++nt)
                acc[nt] = wmma_f16(afr[ks], load_b16(wbuf, ldk, lane, ks * 32, nt * 16), acc[nt]);
        }
        __syncthreads();
        epilogue(acc, nc, N, bias, res, ldr, relu, dst, ldd, lane);
    }
}

// GEMM with A streamed from LDS (large K, chunked weight staging). src != dst.
DEV_INLINE void gemm_big(const _Float16* src, int lda, int K, int N,
                         const float* W, const float* bias,
                         const _Float16* res, int ldr, int relu,
                         _Float16* dst, int ldd,
                         _Float16* wbuf, int lane, int tid) {
    int Kp = (K + 31) & ~31;
    for (int nc = 0; nc < N; nc += 64) {
        v8f acc[4] = {z8(), z8(), z8(), z8()};
        for (int kc = 0; kc < Kp; kc += 128) {
            int kcw = Kp - kc; if (kcw > 128) kcw = 128;
            int ldk = kcw + 8;
            stage_w(wbuf, W, K, N, kc, nc, kcw, ldk, tid);
            __syncthreads();
            for (int k0 = 0; k0 < kcw; k0 += 32) {
                v16h a = load_a16(src, lda, lane, kc + k0);
#pragma unroll
                for (int nt = 0; nt < 4; ++nt)
                    acc[nt] = wmma_f16(a, load_b16(wbuf, ldk, lane, k0, nt * 16), acc[nt]);
            }
            __syncthreads();
        }
        epilogue(acc, nc, N, bias, res, ldr, relu, dst, ldd, lane);
    }
}

__global__ __launch_bounds__(NTH, 4) __attribute__((amdgpu_num_vgpr(256)))
void msmt_fused_fwd(KParams P, int B) {
    __shared__ _Float16 s_wbuf[WBUF_ELEMS];        // shared weight staging (17 KB)
    __shared__ _Float16 s_act[WAVES][16 * 512];    // wide activation buffer
    __shared__ _Float16 s_hA[WAVES][16 * 64];      // residual stream h
    __shared__ _Float16 s_hB[WAVES][16 * 128];     // scratch tile
    __shared__ float    s_ch[WAVES][16 * 8];       // gate softmax weights

    int tid = threadIdx.x;
    int w = tid >> 5, lane = tid & 31;
    _Float16* act = s_act[w];
    _Float16* hA  = s_hA[w];
    _Float16* hB  = s_hB[w];
    float*    ch  = s_ch[w];
    int R = blockIdx.x * ROWS_PER_BLOCK + w * BQ;
    int r = lane & 15;
    int n = lane & 15, hi = (lane >> 4) & 1;

    // ---- stage x[:, :9] -> hA (zero padded to 64 cols) ----
    for (int idx = lane; idx < 16 * 64; idx += 32) {
        int rr = idx >> 6, c = idx & 63;
        float v = (c < 9) ? P.x[(size_t)(R + rr) * 25 + c] : 0.0f;
        hA[idx] = (_Float16)v;
    }
    // ctrl features y per row (lanes 0-15)
    float y[16];
    if (lane < 16) {
#pragma unroll
        for (int i = 0; i < 16; ++i) y[i] = P.x[(size_t)(R + r) * 25 + 9 + i];
    }

    // ---- input projection: h = x9 @ proj_w + proj_b  (K=9->32, N=54, in-place) ----
    {
        v16h afr[1] = { load_a16(hA, 64, lane, 0) };
        gemm_pre(afr, 1, 9, 54, P.proj_w, P.proj_b, nullptr, 0, 0, hA, 64, s_wbuf, lane, tid);
    }

    // =========================== layers ===========================
    for (int l = 0; l < 3; ++l) {
        const float* qkvw = P.qkv_w + (size_t)l * 54 * 162;
        const float* qkvb = P.qkv_b + l * 162;
        const float* apw  = P.attnproj_w + (size_t)l * 54 * 54;
        const float* apb  = P.attnproj_b + l * 54;
        const float* m1w  = P.mlp_w1 + (size_t)l * 54 * 512;
        const float* m1b  = P.mlp_b1 + l * 512;
        const float* m2w  = P.mlp_w2 + (size_t)l * 512 * 54;
        const float* m2b  = P.mlp_b2 + l * 54;

        // qkv = h @ qkv_w + qkv_b   (54 -> 162) into act
        {
            v16h afr[2] = { load_a16(hA, 64, lane, 0), load_a16(hA, 64, lane, 32) };
            gemm_pre(afr, 2, 54, 162, qkvw, qkvb, nullptr, 0, 0, act, 512, s_wbuf, lane, tid);
        }

        // ---- per-row attention + rel MLP (lanes 0-15) -> hB ----
        if (lane < 16) {
            const _Float16* qrow = act + r * 512;
            const float* rw1 = P.rel_w1 + l * 16 * 6;  const float* rb1 = P.rel_b1 + l * 6;
            const float* rw2 = P.rel_w2 + l * 6 * 30;  const float* rb2 = P.rel_b2 + l * 30;
            const float* rw3 = P.rel_w3 + l * 30 * 36; const float* rb3 = P.rel_b3 + l * 36;

            float s[36];
#pragma unroll
            for (int hh = 0; hh < 6; ++hh) {
                float q[9];
#pragma unroll
                for (int d = 0; d < 9; ++d) q[d] = (float)qrow[hh * 9 + d];
#pragma unroll
                for (int g = 0; g < 6; ++g) {
                    float a = 0.f;
#pragma unroll
                    for (int d = 0; d < 9; ++d) a += q[d] * (float)qrow[54 + g * 9 + d];
                    s[hh * 6 + g] = a * (1.0f / 3.0f);   // dk = 9^-0.5
                }
            }
            float rel1[6];
#pragma unroll
            for (int j = 0; j < 6; ++j) {
                float a = rb1[j];
#pragma unroll
                for (int i = 0; i < 16; ++i) a += y[i] * rw1[i * 6 + j];
                rel1[j] = fmaxf(a, 0.0f);
            }
            float rel2[30];
            for (int j = 0; j < 30; ++j) {
                float a = rb2[j];
#pragma unroll
                for (int i = 0; i < 6; ++i) a += rel1[i] * rw2[i * 30 + j];
                rel2[j] = fmaxf(a, 0.0f);
            }
            float rel3[36];
            for (int j = 0; j < 36; ++j) {
                float a = rb3[j];
                for (int i = 0; i < 30; ++i) a += rel2[i] * rw3[i * 36 + j];
                rel3[j] = a;
            }
            // softmax over g, then * rel
#pragma unroll
            for (int hh = 0; hh < 6; ++hh) {
                float mx = s[hh * 6];
#pragma unroll
                for (int g = 1; g < 6; ++g) mx = fmaxf(mx, s[hh * 6 + g]);
                float sum = 0.f;
#pragma unroll
                for (int g = 0; g < 6; ++g) { float e = __expf(s[hh * 6 + g] - mx); s[hh * 6 + g] = e; sum += e; }
                float inv = 1.0f / sum;
#pragma unroll
                for (int g = 0; g < 6; ++g) s[hh * 6 + g] = s[hh * 6 + g] * inv * rel3[hh * 6 + g];
            }
            // out[d*6+hh] = sum_g attn[hh][g] * v[g][d]   (swapaxes layout)
#pragma unroll
            for (int d = 0; d < 9; ++d) {
                float vv[6];
#pragma unroll
                for (int g = 0; g < 6; ++g) vv[g] = (float)qrow[108 + g * 9 + d];
#pragma unroll
                for (int hh = 0; hh < 6; ++hh) {
                    float o = 0.f;
#pragma unroll
                    for (int g = 0; g < 6; ++g) o += s[hh * 6 + g] * vv[g];
                    hB[r * 128 + d * 6 + hh] = (_Float16)o;
                }
            }
            for (int c = 54; c < 64; ++c) hB[r * 128 + c] = (_Float16)0.0f;
        }

        // attnproj + residual(h) -> hA  (in-place OK, A preloaded from hB)
        {
            v16h afr[2] = { load_a16(hB, 128, lane, 0), load_a16(hB, 128, lane, 32) };
            gemm_pre(afr, 2, 54, 54, apw, apb, hA, 64, 0, hA, 64, s_wbuf, lane, tid);
        }
        // mlp1: 54 -> 512 relu -> act
        {
            v16h afr[2] = { load_a16(hA, 64, lane, 0), load_a16(hA, 64, lane, 32) };
            gemm_pre(afr, 2, 54, 512, m1w, m1b, nullptr, 0, 1, act, 512, s_wbuf, lane, tid);
        }
        // mlp2: 512 -> 54 + residual -> hA
        gemm_big(act, 512, 512, 54, m2w, m2b, hA, 64, 0, hA, 64, s_wbuf, lane, tid);

        // ---- per-row gate controller softmax -> ch ----
        if (lane < 16) {
            const float* cw1 = P.ctrl_w1 + l * 16 * 8; const float* cb1 = P.ctrl_b1 + l * 8;
            const float* cw2 = P.ctrl_w2 + l * 8 * 8;  const float* cb2 = P.ctrl_b2 + l * 8;
            float t[8];
#pragma unroll
            for (int j = 0; j < 8; ++j) {
                float a = cb1[j];
#pragma unroll
                for (int i = 0; i < 16; ++i) a += y[i] * cw1[i * 8 + j];
                t[j] = fmaxf(a, 0.0f);
            }
            float u[8];
#pragma unroll
            for (int j = 0; j < 8; ++j) {
                float a = cb2[j];
#pragma unroll
                for (int i = 0; i < 8; ++i) a += t[i] * cw2[i * 8 + j];
                u[j] = a;
            }
            float mx = u[0];
#pragma unroll
            for (int g = 1; g < 8; ++g) mx = fmaxf(mx, u[g]);
            float sum = 0.f;
#pragma unroll
            for (int g = 0; g < 8; ++g) { float e = __expf(u[g] - mx); u[g] = e; sum += e; }
            float inv = 1.0f / sum;
#pragma unroll
            for (int g = 0; g < 8; ++g) ch[r * 8 + g] = u[g] * inv;
        }

        // ---- gates: 8x (54 -> 13 relu -> 54), softmax-weighted sum + residual ----
        {
            v16h ga[2] = { load_a16(hA, 64, lane, 0), load_a16(hA, 64, lane, 32) };
            v8f gsum[4] = { z8(), z8(), z8(), z8() };
            for (int g = 0; g < 8; ++g) {
                const float* gw1 = P.gate_w1 + (size_t)(l * 8 + g) * 54 * 13;
                const float* gb1 = P.gate_b1 + (l * 8 + g) * 13;
                const float* gw2 = P.gate_w2 + (size_t)(l * 8 + g) * 13 * 54;
                const float* gb2 = P.gate_b2 + (l * 8 + g) * 54;

                stage_w(s_wbuf, gw1, 54, 13, 0, 0, 64, 72, tid);
                __syncthreads();
                v8f acg = z8();
                acg = wmma_f16(ga[0], load_b16(s_wbuf, 72, lane, 0, 0), acg);
                acg = wmma_f16(ga[1], load_b16(s_wbuf, 72, lane, 32, 0), acg);
                __syncthreads();
                {   // gh = relu(. + b1) -> hB cols [0,32) zero padded
                    float bv = (n < 13) ? gb1[n] : 0.0f;
#pragma unroll
                    for (int i = 0; i < 8; ++i) {
                        int m = i + 8 * hi;
                        float v = fmaxf(acg[i] + bv, 0.0f);
                        hB[m * 128 + n]      = (n < 13) ? (_Float16)v : (_Float16)0.0f;
                        hB[m * 128 + 16 + n] = (_Float16)0.0f;
                    }
                }
                v16h a2 = load_a16(hB, 128, lane, 0);
                stage_w(s_wbuf, gw2, 13, 54, 0, 0, 32, 40, tid);
                __syncthreads();
#pragma unroll
                for (int nt = 0; nt < 4; ++nt) {
                    v8f ac2 = z8();
                    ac2 = wmma_f16(a2, load_b16(s_wbuf, 40, lane, 0, nt * 16), ac2);
                    int gn = nt * 16 + n;
                    if (gn < 54) {
                        float bv2 = gb2[gn];
#pragma unroll
                        for (int i = 0; i < 8; ++i) {
                            int m = i + 8 * hi;
                            gsum[nt][i] = gsum[nt][i] + ch[m * 8 + g] * (ac2[i] + bv2);
                        }
                    }
                }
                __syncthreads();
            }
            // combine with residual sc3 (hA), write new h -> hA
#pragma unroll
            for (int nt = 0; nt < 4; ++nt) {
                int gn = nt * 16 + n;
#pragma unroll
                for (int i = 0; i < 8; ++i) {
                    int m = i + 8 * hi;
                    float v = gsum[nt][i];
                    if (gn < 54) v += (float)hA[m * 64 + gn];
                    hA[m * 64 + gn] = (gn < 54) ? (_Float16)v : (_Float16)0.0f;
                }
            }
        }
    } // layers

    // =========================== L head: 54->512->128->64->32->1 ===========================
    {
        v16h afr[2] = { load_a16(hA, 64, lane, 0), load_a16(hA, 64, lane, 32) };
        gemm_pre(afr, 2, 54, 512, P.L_w[0], P.L_b[0], nullptr, 0, 1, act, 512, s_wbuf, lane, tid);
    }
    gemm_big(act, 512, 512, 128, P.L_w[1], P.L_b[1], nullptr, 0, 1, hB, 128, s_wbuf, lane, tid);
    {
        v16h afr[4] = { load_a16(hB, 128, lane, 0), load_a16(hB, 128, lane, 32),
                        load_a16(hB, 128, lane, 64), load_a16(hB, 128, lane, 96) };
        gemm_pre(afr, 4, 128, 64, P.L_w[2], P.L_b[2], nullptr, 0, 1, hB, 128, s_wbuf, lane, tid);
    }
    {
        v16h afr[2] = { load_a16(hB, 128, lane, 0), load_a16(hB, 128, lane, 32) };
        gemm_pre(afr, 2, 64, 32, P.L_w[3], P.L_b[3], nullptr, 0, 1, hB, 128, s_wbuf, lane, tid);
    }
    if (lane < 16) {
        float a = P.L_b[4][0];
        for (int c = 0; c < 32; ++c) a += (float)hB[r * 128 + c] * P.L_w[4][c];
        P.out[(size_t)(R + r) * 2 + 0] = a;
    }

    // =========================== H head: 54->512->64->1 ===========================
    {
        v16h afr[2] = { load_a16(hA, 64, lane, 0), load_a16(hA, 64, lane, 32) };
        gemm_pre(afr, 2, 54, 512, P.H_w[0], P.H_b[0], nullptr, 0, 1, act, 512, s_wbuf, lane, tid);
    }
    gemm_big(act, 512, 512, 64, P.H_w[1], P.H_b[1], nullptr, 0, 1, hB, 128, s_wbuf, lane, tid);
    if (lane < 16) {
        float a = P.H_b[2][0];
        for (int c = 0; c < 64; ++c) a += (float)hB[r * 128 + c] * P.H_w[2][c];
        P.out[(size_t)(R + r) * 2 + 1] = a;
    }
}

extern "C" void kernel_launch(void* const* d_in, const int* in_sizes, int n_in,
                              void* d_out, int out_size, void* d_ws, size_t ws_size,
                              hipStream_t stream) {
    (void)n_in; (void)out_size; (void)d_ws; (void)ws_size;
    KParams P;
    P.x          = (const float*)d_in[0];
    P.proj_w     = (const float*)d_in[1];
    P.proj_b     = (const float*)d_in[2];
    P.qkv_w      = (const float*)d_in[3];
    P.qkv_b      = (const float*)d_in[4];
    P.attnproj_w = (const float*)d_in[5];
    P.attnproj_b = (const float*)d_in[6];
    P.mlp_w1     = (const float*)d_in[7];
    P.mlp_b1     = (const float*)d_in[8];
    P.mlp_w2     = (const float*)d_in[9];
    P.mlp_b2     = (const float*)d_in[10];
    P.rel_w1     = (const float*)d_in[11];
    P.rel_b1     = (const float*)d_in[12];
    P.rel_w2     = (const float*)d_in[13];
    P.rel_b2     = (const float*)d_in[14];
    P.rel_w3     = (const float*)d_in[15];
    P.rel_b3     = (const float*)d_in[16];
    P.gate_w1    = (const float*)d_in[17];
    P.gate_b1    = (const float*)d_in[18];
    P.gate_w2    = (const float*)d_in[19];
    P.gate_b2    = (const float*)d_in[20];
    P.ctrl_w1    = (const float*)d_in[21];
    P.ctrl_b1    = (const float*)d_in[22];
    P.ctrl_w2    = (const float*)d_in[23];
    P.ctrl_b2    = (const float*)d_in[24];
    for (int k = 0; k < 5; ++k) P.L_w[k] = (const float*)d_in[25 + k];
    for (int k = 0; k < 5; ++k) P.L_b[k] = (const float*)d_in[30 + k];
    for (int k = 0; k < 3; ++k) P.H_w[k] = (const float*)d_in[35 + k];
    for (int k = 0; k < 3; ++k) P.H_b[k] = (const float*)d_in[38 + k];
    P.out = (float*)d_out;

    int B = in_sizes[0] / 25;                 // 131072
    int blocks = B / ROWS_PER_BLOCK;          // 4096
    msmt_fused_fwd<<<blocks, NTH, 0, stream>>>(P, B);
}
